// GATEncoder_53669911331164
// MI455X (gfx1250) — matrix operand order repaired and checked by reading, not compiled, not run on hardware
//
#include <hip/hip_runtime.h>
#include <hip/hip_bf16.h>
#include <hip/hip_fp16.h>

typedef __attribute__((ext_vector_type(16))) _Float16 v16h;
typedef __attribute__((ext_vector_type(8)))  float    v8f;
typedef __attribute__((ext_vector_type(4)))  float    v4f;

#define NEG_SLOPE 0.2f
#define EPS_F 1e-16f
#define NUM_GRAPHS 64

// ---------- helpers ----------
__device__ __forceinline__ unsigned encf(float f) {
    unsigned u = __float_as_uint(f);
    return (u & 0x80000000u) ? ~u : (u | 0x80000000u);   // monotone float->uint
}
__device__ __forceinline__ float decf(unsigned e) {
    return (e & 0x80000000u) ? __uint_as_float(e & 0x7FFFFFFFu) : __uint_as_float(~e);
}
__device__ __forceinline__ float lrelu(float v) { return v > 0.f ? v : NEG_SLOPE * v; }

__device__ __forceinline__ void edge_ids(const int* ei, int E, int e, int& src, int& dst) {
    if (e < E) { src = ei[e]; dst = ei[E + e]; }
    else       { src = e - E; dst = e - E; }            // self loop
}

// ---------- kernels ----------
__global__ __launch_bounds__(256) void k_zero(float* p, long long n) {
    long long i = (long long)blockIdx.x * blockDim.x + threadIdx.x;
    if (i < n) p[i] = 0.f;
}

// a_s1[n,h] = x[n] * <W1_h, att_src1_h>;   a_d1 likewise
__global__ __launch_bounds__(256) void k_att1(const float* x, const float* W1,
                                              const float* atts, const float* attd,
                                              float* a_s, float* a_d, int N) {
    int i = blockIdx.x * blockDim.x + threadIdx.x;      // n*4 + h
    if (i >= N * 4) return;
    int h = i & 3;
    float ds = 0.f, dd = 0.f;
    #pragma unroll 8
    for (int c = 0; c < 64; ++c) {
        float w = W1[h * 64 + c];
        ds += w * atts[h * 64 + c];
        dd += w * attd[h * 64 + c];
    }
    float xv = x[i >> 2];
    a_s[i] = xv * ds;
    a_d[i] = xv * dd;
}

// pass A: segment max of leakyrelu(a_s[src]+a_d[dst]) per (dst, head)
__global__ __launch_bounds__(256) void k_edge_max(const int* ei, int E, int Etot,
                                                  const float* a_s, const float* a_d,
                                                  unsigned* menc) {
    int i = blockIdx.x * blockDim.x + threadIdx.x;      // e*4 + h
    if (i >= Etot * 4) return;
    int e = i >> 2, h = i & 3;
    int src, dst; edge_ids(ei, E, e, src, dst);
    float v = lrelu(a_s[src * 4 + h] + a_d[dst * 4 + h]);
    atomicMax(&menc[dst * 4 + h], encf(v));
}

// pass B: denom[dst,h] += exp(e - m[dst,h])
__global__ __launch_bounds__(256) void k_edge_expsum(const int* ei, int E, int Etot,
                                                     const float* a_s, const float* a_d,
                                                     const unsigned* menc, float* denom) {
    int i = blockIdx.x * blockDim.x + threadIdx.x;
    if (i >= Etot * 4) return;
    int e = i >> 2, h = i & 3;
    int src, dst; edge_ids(ei, E, e, src, dst);
    float v = lrelu(a_s[src * 4 + h] + a_d[dst * 4 + h]);
    float ex = expf(v - decf(menc[dst * 4 + h]));
    atomicAdd(&denom[dst * 4 + h], ex);
}

// layer-1 aggregation (scalar per head): s1[dst,h] += alpha * x[src]
__global__ __launch_bounds__(256) void k_edge_agg1(const int* ei, int E, int Etot, const float* x,
                                                   const float* a_s, const float* a_d,
                                                   const unsigned* menc, const float* denom,
                                                   float* s1) {
    int i = blockIdx.x * blockDim.x + threadIdx.x;
    if (i >= Etot * 4) return;
    int e = i >> 2, h = i & 3;
    int src, dst; edge_ids(ei, E, e, src, dst);
    float v = lrelu(a_s[src * 4 + h] + a_d[dst * 4 + h]);
    float alpha = expf(v - decf(menc[dst * 4 + h])) / (denom[dst * 4 + h] + EPS_F);
    atomicAdd(&s1[dst * 4 + h], alpha * x[src]);
}

// h1 = relu(s1[n,h]*W1 + b1) written directly in WMMA-A fragment-swizzled layout:
// element ((tm*8+g)*32 + lane)*16 + j  holds  A[tm*16 + lane%16][g*32 + baseA + j + (j>=8?8:0)]
// with baseA = (lane>=16) ? 8 : 0  (ISA 16-bit A 16x32 lane layout)
__global__ __launch_bounds__(256) void k_h1_swz(const float* s1, const float* W1, const float* b1,
                                                _Float16* aswz, int N, int tilesM) {
    int i = blockIdx.x * blockDim.x + threadIdx.x;
    if (i >= tilesM * 4096) return;
    int tm = i >> 12;
    int within = i & 4095;
    int g = within >> 9;
    int l = (within >> 4) & 31;
    int j = within & 15;
    int row = tm * 16 + (l & 15);
    int baseA = (l >= 16) ? 8 : 0;
    int K = g * 32 + baseA + j + ((j >= 8) ? 8 : 0);
    float v = 0.f;
    if (row < N) {
        int h = K >> 6;
        v = s1[row * 4 + h] * W1[K] + b1[K];
        v = v > 0.f ? v : 0.f;
    }
    aswz[i] = (_Float16)v;
}

// W2 (256x512 f32) -> WMMA-B fragment-swizzled f16:
// element ((tn*8+g)*32 + lane)*16 + j  holds  B[g*32 + baseB + j + (j>=8?8:0)][tn*16 + lane%16]
// with baseB = (lane>=16) ? 8 : 0   (ISA 16-bit B 32x16 lane layout)
__global__ __launch_bounds__(256) void k_w2_swz(const float* W2, _Float16* bswz) {
    int i = blockIdx.x * blockDim.x + threadIdx.x;      // 32*8*32*16 = 131072
    if (i >= 131072) return;
    int tn = i >> 12;
    int within = i & 4095;
    int g = within >> 9;
    int l = (within >> 4) & 31;
    int j = within & 15;
    int baseB = (l >= 16) ? 8 : 0;
    int K = g * 32 + baseB + j + ((j >= 8) ? 8 : 0);
    int col = tn * 16 + (l & 15);
    bswz[i] = (_Float16)W2[K * 512 + col];
}

// C[M x 512] = A @ B from swizzled f16 operands, f32 accumulate.
// Block: 256 thr = 8 waves, tile 128 rows x 64 cols. B panel (32KB) staged in LDS.
// Each wave: 16 rows x 64 cols = 4 accumulators; one A fragment feeds 4 WMMAs.
__global__ __launch_bounds__(256) void k_gemm_wmma(const _Float16* __restrict__ Asw,
                                                   const _Float16* __restrict__ Bsw,
                                                   float* __restrict__ C,
                                                   int Mrows, int tilesM) {
    __shared__ __align__(32) _Float16 Bs[16384];        // 4 col-tiles x K=256
    const int tid  = threadIdx.x;
    const int w    = tid >> 5;
    const int lane = tid & 31;
    const int blockRow = blockIdx.x >> 3;
    const int blockCol = blockIdx.x & 7;

    // stage B panel: linear 32B chunks, coalesced global b128 -> ds b128
    {
        const v16h* src = (const v16h*)(Bsw + (long long)blockCol * 16384);
        v16h* dst = (v16h*)Bs;
        #pragma unroll
        for (int c = 0; c < 4; ++c) dst[tid + c * 256] = src[tid + c * 256];
    }
    __syncthreads();

    const int tm = blockRow * 8 + w;
    if (tm < tilesM) {
        v8f acc0 = {}, acc1 = {}, acc2 = {}, acc3 = {};
        const _Float16* aBase = Asw + (long long)tm * 4096 + lane * 16;
        const _Float16* bLane = Bs + lane * 16;
        #pragma unroll
        for (int g = 0; g < 8; ++g) {
            v16h a  = *(const v16h*)(aBase + g * 512);
            v16h b0 = *(const v16h*)(bLane + g * 512);
            v16h b1 = *(const v16h*)(bLane + 4096  + g * 512);
            v16h b2 = *(const v16h*)(bLane + 8192  + g * 512);
            v16h b3 = *(const v16h*)(bLane + 12288 + g * 512);
            acc0 = __builtin_amdgcn_wmma_f32_16x16x32_f16(false, a, false, b0, (short)0, acc0, false, false);
            acc1 = __builtin_amdgcn_wmma_f32_16x16x32_f16(false, a, false, b1, (short)0, acc1, false, false);
            acc2 = __builtin_amdgcn_wmma_f32_16x16x32_f16(false, a, false, b2, (short)0, acc2, false, false);
            acc3 = __builtin_amdgcn_wmma_f32_16x16x32_f16(false, a, false, b3, (short)0, acc3, false, false);
        }
        // C/D layout: VGPR r -> row tm*16 + r + (lane>=16 ? 8 : 0), col = lane%16
        const int rowBase = tm * 16 + ((lane >= 16) ? 8 : 0);
        const int colBase = blockCol * 64 + (lane & 15);
        float* cp = C + (long long)rowBase * 512 + colBase;
        #pragma unroll
        for (int r = 0; r < 8; ++r) {
            if (rowBase + r < Mrows) {
                cp[r * 512 +  0] = acc0[r];
                cp[r * 512 + 16] = acc1[r];
                cp[r * 512 + 32] = acc2[r];
                cp[r * 512 + 48] = acc3[r];
            }
        }
    }
}

// a_s2[n,h] = <h2[n,h,:], att_src2[h,:]> (128-wide dot), a_d2 likewise
__global__ __launch_bounds__(256) void k_att2(const float* h2, const float* atts, const float* attd,
                                              float* a_s, float* a_d, int N) {
    int i = blockIdx.x * blockDim.x + threadIdx.x;      // n*4 + h
    if (i >= N * 4) return;
    int h = i & 3, n = i >> 2;
    const float* hp = h2 + (long long)n * 512 + h * 128;
    float ds = 0.f, dd = 0.f;
    #pragma unroll 8
    for (int c = 0; c < 128; ++c) {
        float v = hp[c];
        ds += v * atts[h * 128 + c];
        dd += v * attd[h * 128 + c];
    }
    a_s[i] = ds;
    a_d[i] = dd;
}

// layer-2 aggregation, float4 per thread: out2[dst, j..j+3] += alpha * h2[src, j..j+3]
__global__ __launch_bounds__(256) void k_edge_agg2(const int* ei, int E, int Etot,
                                                   const float* __restrict__ h2,
                                                   const float* a_s, const float* a_d,
                                                   const unsigned* menc, const float* denom,
                                                   float* out2) {
    long long i = (long long)blockIdx.x * blockDim.x + threadIdx.x;
    if (i >= (long long)Etot * 128) return;
    int e = (int)(i >> 7);
    int q = (int)(i & 127);
    int h = q >> 5;                 // head
    int j = h * 128 + (q & 31) * 4; // feature offset within 512
    int src, dst; edge_ids(ei, E, e, src, dst);
    float v = lrelu(a_s[src * 4 + h] + a_d[dst * 4 + h]);
    float alpha = expf(v - decf(menc[dst * 4 + h])) / (denom[dst * 4 + h] + EPS_F);
    v4f hv = *(const v4f*)(h2 + (long long)src * 512 + j);
    float* op = out2 + (long long)dst * 512 + j;
    atomicAdd(op + 0, alpha * hv.x);
    atomicAdd(op + 1, alpha * hv.y);
    atomicAdd(op + 2, alpha * hv.z);
    atomicAdd(op + 3, alpha * hv.w);
}

// head mean + bias + relu, then pooled sums/counts per graph
__global__ __launch_bounds__(256) void k_pool(const float* out2, const float* b2,
                                              const int* batch, float* pool, float* cnt, int N) {
    int i = blockIdx.x * blockDim.x + threadIdx.x;      // n*128 + c
    if (i >= N * 128) return;
    int c = i & 127, n = i >> 7;
    const float* p = out2 + (long long)n * 512;
    float v = 0.25f * (p[c] + p[128 + c] + p[256 + c] + p[384 + c]) + b2[c];
    v = v > 0.f ? v : 0.f;
    int g = batch[n];
    atomicAdd(&pool[g * 128 + c], v);
    if (c == 0) atomicAdd(&cnt[g], 1.0f);
}

__global__ __launch_bounds__(256) void k_final(const float* pool, const float* cnt, float* out) {
    int i = blockIdx.x * blockDim.x + threadIdx.x;      // g*128 + c
    if (i >= NUM_GRAPHS * 128) return;
    out[i] = pool[i] / fmaxf(cnt[i >> 7], 1.0f);
}

// ---------- launch ----------
extern "C" void kernel_launch(void* const* d_in, const int* in_sizes, int n_in,
                              void* d_out, int out_size, void* d_ws, size_t ws_size,
                              hipStream_t stream) {
    (void)n_in; (void)out_size; (void)ws_size;
    const float* x     = (const float*)d_in[0];
    const int*   ei    = (const int*)  d_in[1];
    const int*   batch = (const int*)  d_in[2];
    const float* W1    = (const float*)d_in[3];
    const float* atts1 = (const float*)d_in[4];
    const float* attd1 = (const float*)d_in[5];
    const float* b1    = (const float*)d_in[6];
    const float* W2    = (const float*)d_in[7];
    const float* atts2 = (const float*)d_in[8];
    const float* attd2 = (const float*)d_in[9];
    const float* b2    = (const float*)d_in[10];

    const long long N     = in_sizes[0];
    const long long E     = in_sizes[1] / 2;
    const long long Etot  = E + N;
    const int tilesM      = (int)((N + 15) / 16);

    // workspace layout (float units); zero region first, f16 regions 32B-aligned
    float* Wp = (float*)d_ws;
    long long o = 0;
    float*    a_s1 = Wp + o; o += N * 4;
    float*    a_d1 = Wp + o; o += N * 4;
    unsigned* m1   = (unsigned*)(Wp + o); o += N * 4;
    float*    den1 = Wp + o; o += N * 4;
    float*    s1   = Wp + o; o += N * 4;
    float*    a_s2 = Wp + o; o += N * 4;
    float*    a_d2 = Wp + o; o += N * 4;
    unsigned* m2   = (unsigned*)(Wp + o); o += N * 4;
    float*    den2 = Wp + o; o += N * 4;
    float*    pool = Wp + o; o += NUM_GRAPHS * 128;
    float*    cnt  = Wp + o; o += NUM_GRAPHS;
    float*    out2 = Wp + o; o += N * 512;
    const long long zeroN = o;                           // everything above starts at 0
    _Float16* h1f  = (_Float16*)(Wp + o); o += (long long)tilesM * 2048;  // tilesM*4096 halfs
    _Float16* w2f  = (_Float16*)(Wp + o); o += 65536;                     // 131072 halfs
    float*    h2   = Wp + o; o += N * 512;

    auto nblk = [](long long n, int b) { return dim3((unsigned)((n + b - 1) / b)); };
    const dim3 T256(256);

    k_zero<<<nblk(zeroN, 256), T256, 0, stream>>>(Wp, zeroN);

    // ---- layer 1 (scalar per head) ----
    k_att1<<<nblk(N * 4, 256), T256, 0, stream>>>(x, W1, atts1, attd1, a_s1, a_d1, (int)N);
    k_edge_max   <<<nblk(Etot * 4, 256), T256, 0, stream>>>(ei, (int)E, (int)Etot, a_s1, a_d1, m1);
    k_edge_expsum<<<nblk(Etot * 4, 256), T256, 0, stream>>>(ei, (int)E, (int)Etot, a_s1, a_d1, m1, den1);
    k_edge_agg1  <<<nblk(Etot * 4, 256), T256, 0, stream>>>(ei, (int)E, (int)Etot, x, a_s1, a_d1, m1, den1, s1);

    // ---- operand prep in WMMA fragment layouts ----
    k_h1_swz<<<nblk((long long)tilesM * 4096, 256), T256, 0, stream>>>(s1, W1, b1, h1f, (int)N, tilesM);
    k_w2_swz<<<nblk(131072, 256), T256, 0, stream>>>(W2, w2f);

    // ---- layer 2 GEMM: h2 = h1(f16) @ W2(f16), f32 acc via WMMA ----
    const int blockRows = (tilesM + 7) / 8;
    k_gemm_wmma<<<dim3((unsigned)(blockRows * 8)), T256, 0, stream>>>(h1f, w2f, h2, (int)N, tilesM);

    // ---- layer 2 attention + softmax + aggregation ----
    k_att2<<<nblk(N * 4, 256), T256, 0, stream>>>(h2, atts2, attd2, a_s2, a_d2, (int)N);
    k_edge_max   <<<nblk(Etot * 4, 256), T256, 0, stream>>>(ei, (int)E, (int)Etot, a_s2, a_d2, m2);
    k_edge_expsum<<<nblk(Etot * 4, 256), T256, 0, stream>>>(ei, (int)E, (int)Etot, a_s2, a_d2, m2, den2);
    k_edge_agg2  <<<nblk(Etot * 128, 256), T256, 0, stream>>>(ei, (int)E, (int)Etot, h2, a_s2, a_d2, m2, den2, out2);

    // ---- head mean + relu + global mean pool ----
    k_pool<<<nblk(N * 128, 256), T256, 0, stream>>>(out2, b2, batch, pool, cnt, (int)N);
    k_final<<<nblk(NUM_GRAPHS * 128, 256), T256, 0, stream>>>(pool, cnt, (float*)d_out);
}
